// SFN_61091614818821
// MI455X (gfx1250) — matrix operand order repaired and checked by reading, not compiled, not run on hardware
//
#include <hip/hip_runtime.h>
#include <hip/hip_bf16.h>

typedef __attribute__((ext_vector_type(16))) _Float16 v16h;
typedef __attribute__((ext_vector_type(8)))  _Float16 v8h;
typedef __attribute__((ext_vector_type(8)))  float    v8f;

#if defined(__gfx1250__) && __has_builtin(__builtin_amdgcn_global_load_async_to_lds_b128) && __has_builtin(__builtin_amdgcn_s_wait_asynccnt)
#define QC_ASYNC 1
typedef int qc_v4i __attribute__((vector_size(16)));   // matches builtin's param type
#endif

// WAGE 8-bit quantizer: round to 1/128 grid, clip to +-127/128.
__device__ __forceinline__ float wage_q8(float x) {
  float q = rintf(x * 128.0f) * 0.0078125f;
  return fminf(fmaxf(q, -0.9921875f), 0.9921875f);
}

// ---------------------------------------------------------------------------
// Weight pre-quantization + packing:
//   fp32 OIHW (Co x Ktot) -> wage_q8 -> f16 packed as [k_block][co][32],
//   K padded to multiple of 32 with zeros. Each K-step's 32co x 32k A tile is
//   then one contiguous 2KB 64B-aligned block (DMA-friendly).
// ---------------------------------------------------------------------------
__global__ void quant_w_pack_kernel(const float* __restrict__ w,
                                    _Float16* __restrict__ wqp,
                                    int Co, int Kt, int Kpad) {
  int i = blockIdx.x * blockDim.x + threadIdx.x;
  if (i >= Co * Kpad) return;
  int co = i / Kpad, k = i - co * Kpad;
  _Float16 v = (_Float16)0.0f;
  if (k < Kt) v = (_Float16)wage_q8(w[(long)co * Kt + k]);
  wqp[(((long)(k >> 5)) * Co + co) * 32 + (k & 31)] = v;
}

// ---------------------------------------------------------------------------
// Implicit-GEMM quantized conv via v_wmma_f32_16x16x32_f16.
//   GEMM view: M = Co, N = B*Ho*Wo, K = Ci*Kh*Kw (K stepped by 32).
//   Block = 256 threads = 8 waves. Block tile 32 Co x 256 N.
//   Each wave: 16x64 D tile = 4 accumulators -> 4 WMMAs per K-step,
//   one A fragment reused across 4 B fragments.
//   A tile staged by async global->LDS DMA (waves 0-3, one
//   global_load_async_to_lds_b128 each, overlapped with the im2col B fill,
//   completed with s_wait_asynccnt). B tile staged via register gather.
//   As/Bs rows padded to 40 halves (80B): fragment reads are 16B-aligned
//   ds_load_b128 with a 20-bank stride (conflict-free, wave32).
// Epilogue: ep==0 -> out = q(acc); ep==1 -> out = (1 - q(acc)) * res (gate).
// Explicit pad_hw folded into `pad`; crop_like folded into Ho/Wo.
// ---------------------------------------------------------------------------
__global__ __launch_bounds__(256)
void qconv_wmma_kernel(const void* __restrict__ xin, int x_is_f32,
                       const _Float16* __restrict__ wq,   // packed [kb][co][32]
                       const _Float16* __restrict__ res,
                       _Float16* __restrict__ out,
                       int Ci, int Hi, int Wi,
                       int Co, int Kh, int Kw, int stride, int pad,
                       int Ho, int Wo, int epmode) {
  __shared__ _Float16 As[32][40];    // [co_local][k_local], stride 80B
  __shared__ _Float16 Bs[256][40];   // [n_local][k_local],  stride 80B

  const int tid  = threadIdx.x;
  const int lane = tid & 31;
  const int wave = tid >> 5;
  const int b    = blockIdx.z;
  const int coBlock = blockIdx.x * 32;
  const int nBlock  = blockIdx.y * 256;
  const int Ntot = Ho * Wo;
  const int KK   = Kh * Kw;
  const int Ktot = Ci * KK;

  const int mWave = coBlock + ((wave >> 2) << 4);   // wave's Co base
  const int nWave = nBlock  + ((wave & 3) << 6);    // wave's N base (64 wide)

  // ---- B staging setup: this thread owns output position n = nBlock+tid ----
  const int  ng = nBlock + tid;
  const bool nv = ng < Ntot;
  const int  oh = nv ? ng / Wo : 0;
  const int  ow = nv ? ng - oh * Wo : 0;
  const int  ihB = oh * stride - pad;
  const int  iwB = ow * stride - pad;

  const float*    xf = (const float*)xin;
  const _Float16* xh = (const _Float16*)xin;

  v8f acc0 = {}, acc1 = {}, acc2 = {}, acc3 = {};

  // incremental (ci,kh,kw) decomposition of k (uniform across threads -> SALU)
  int ci0 = 0, kh0 = 0, kw0 = 0;

  for (int k0 = 0; k0 < Ktot; k0 += 32) {
    // ---- stage A tile (contiguous 2KB of packed weights) ----
    // chunk c = tid (0..127): 16B from tile+16c -> As[c>>2][(c&3)*8]
    const _Float16* tile = wq + ((long)(k0 >> 5) * Co + coBlock) * 32;
    if (tid < 128) {
      const _Float16* gsrc = tile + tid * 8;
      _Float16*       ldst = &As[tid >> 2][(tid & 3) * 8];
#ifdef QC_ASYNC
      __builtin_amdgcn_global_load_async_to_lds_b128(
          (qc_v4i*)(void*)const_cast<_Float16*>(gsrc),
          (qc_v4i*)(void*)ldst,
          0, 0);
#else
      *(v8h*)ldst = *(const v8h*)gsrc;
#endif
    }

    // ---- stage B tile: 32 K rows of this thread's im2col column ----
    // (overlaps the async weight DMA issued above)
    {
      int ci = ci0, kh = kh0, kw = kw0;
#pragma unroll 4
      for (int kk = 0; kk < 32; ++kk) {
        int k = k0 + kk;
        float v = 0.0f;
        if (nv && k < Ktot) {
          int ih = ihB + kh, iw = iwB + kw;
          if (ih >= 0 && ih < Hi && iw >= 0 && iw < Wi) {
            long idx = (((long)b * Ci + ci) * Hi + ih) * Wi + iw;
            v = x_is_f32 ? xf[idx] : (float)xh[idx];
          }
        }
        Bs[tid][kk] = (_Float16)v;
        if (++kw == Kw) { kw = 0; if (++kh == Kh) { kh = 0; ++ci; } }
      }
      ci0 = ci; kh0 = kh; kw0 = kw;
    }

#ifdef QC_ASYNC
    if (tid < 128) __builtin_amdgcn_s_wait_asynccnt(0);
#endif
    __syncthreads();

    // ---- fragments via vector LDS reads (ds_load_b128) ----
    // A 16x32 layout: lane<16 -> K {0..7,16..23}; lane>=16 -> K {8..15,24..31}
    // B 32x16 layout: lane<16 -> K {0..7,16..23}; lane>=16 -> K {8..15,24..31}
    const int hi   = lane >> 4;
    const int mRow = ((wave >> 2) << 4) + (lane & 15);
    const int nRow = ((wave & 3) << 6) + (lane & 15);
    union { v16h v; v8h h[2]; } af;
    af.h[0] = *(const v8h*)&As[mRow][hi << 3];
    af.h[1] = *(const v8h*)&As[mRow][16 + (hi << 3)];

#define QC_WMMA(J, ACC)                                                        \
    { union { v16h v; v8h h[2]; } bf;                                          \
      bf.h[0] = *(const v8h*)&Bs[nRow + (J)*16][hi << 3];                      \
      bf.h[1] = *(const v8h*)&Bs[nRow + (J)*16][16 + (hi << 3)];               \
      ACC = __builtin_amdgcn_wmma_f32_16x16x32_f16(                            \
          false, af.v, false, bf.v, (short)0, ACC, false, false); }

    QC_WMMA(0, acc0)
    QC_WMMA(1, acc1)
    QC_WMMA(2, acc2)
    QC_WMMA(3, acc3)
#undef QC_WMMA
    __syncthreads();
  }

  // ---- epilogue: D layout VGPR r -> M = r + (lane>=16 ? 8:0), N = lane&15 ----
  auto storeEl = [&](float a, int j, int r) {
    int co = mWave + ((lane >> 4) << 3) + r;
    int n  = nWave + j * 16 + (lane & 15);
    if (co < Co && n < Ntot) {
      int oh2 = n / Wo, ow2 = n - oh2 * Wo;
      long oidx = (((long)b * Co + co) * Ho + oh2) * Wo + ow2;
      float y = wage_q8(a);
      if (epmode == 1) y = (1.0f - y) * (float)res[oidx];
      out[oidx] = (_Float16)y;
    }
  };
#pragma unroll
  for (int r = 0; r < 8; ++r) {
    storeEl(acc0[r], 0, r);
    storeEl(acc1[r], 1, r);
    storeEl(acc2[r], 2, r);
    storeEl(acc3[r], 3, r);
  }
}

// ---------------------------------------------------------------------------
// Channel concat of 3 NCHW f16 tensors (same B,H,W).
// ---------------------------------------------------------------------------
__global__ void concat3_kernel(const _Float16* __restrict__ a, int ca,
                               const _Float16* __restrict__ b2, int cb,
                               const _Float16* __restrict__ c, int cc,
                               _Float16* __restrict__ out, int B, int HW) {
  int Cs = ca + cb + cc;
  long total = (long)B * Cs * HW;
  long i = (long)blockIdx.x * blockDim.x + threadIdx.x;
  if (i >= total) return;
  long bc = i / HW;
  int  p  = (int)(i - bc * HW);
  int  ch = (int)(bc % Cs);
  int  bb = (int)(bc / Cs);
  _Float16 v;
  if (ch < ca)            v = a [((long)bb * ca +  ch           ) * HW + p];
  else if (ch < ca + cb)  v = b2[((long)bb * cb + (ch - ca)     ) * HW + p];
  else                    v = c [((long)bb * cc + (ch - ca - cb)) * HW + p];
  out[i] = v;
}

// ---------------------------------------------------------------------------
// Global max-pool: one block per (b,c) plane.
// ---------------------------------------------------------------------------
__global__ void maxpool_kernel(const _Float16* __restrict__ x,
                               float* __restrict__ feat, int HW) {
  __shared__ float red[256];
  long base = (long)blockIdx.x * HW;
  float m = -3.0e38f;
  for (int i = threadIdx.x; i < HW; i += 256)
    m = fmaxf(m, (float)x[base + i]);
  red[threadIdx.x] = m;
  __syncthreads();
  for (int s = 128; s > 0; s >>= 1) {
    if (threadIdx.x < s)
      red[threadIdx.x] = fmaxf(red[threadIdx.x], red[threadIdx.x + s]);
    __syncthreads();
  }
  if (threadIdx.x == 0) feat[blockIdx.x] = red[0];
}

// ---------------------------------------------------------------------------
// QLinear 32 -> 1 on 8 batch rows.
// ---------------------------------------------------------------------------
__global__ void fc_kernel(const float* __restrict__ feat,
                          const float* __restrict__ wfc,
                          float* __restrict__ out) {
  int b = threadIdx.x;
  if (b < 8) {
    float s = 0.0f;
#pragma unroll
    for (int c = 0; c < 32; ++c) s += feat[b * 32 + c] * wage_q8(wfc[c]);
    out[b] = wage_q8(s);
  }
}

// ---------------------------------------------------------------------------
extern "C" void kernel_launch(void* const* d_in, const int* in_sizes, int n_in,
                              void* d_out, int out_size, void* d_ws, size_t ws_size,
                              hipStream_t stream) {
  (void)n_in; (void)out_size; (void)ws_size; (void)in_sizes;
  const float* x = (const float*)d_in[0];
  const float* w[16];
  for (int i = 0; i < 16; ++i) w[i] = (const float*)d_in[i + 1];
  // index: 0 w1  1 w2  2 w3  3 w4  4 wr11 5 wr12 6 wr21 7 wr22
  //        8 wd3 9 wd2 10 wd1 11 wu43 12 wu32 13 wu21 14 wu10 15 wfc

  char* p = (char*)d_ws;
  auto allocH = [&](size_t n) { _Float16* r = (_Float16*)p; p += ((n * 2 + 255) / 256) * 256; return r; };
  auto allocF = [&](size_t n) { float*    r = (float*)p;    p += ((n * 4 + 255) / 256) * 256; return r; };

  // (Co, Ktot) per conv weight; Kpad = Ktot rounded up to 32.
  const int wCo[15] = { 64, 128, 256, 512, 512, 512, 512, 512, 128,  64,   4,  32,  32,  32,  32 };
  const int wKt[15] = { 36, 576, 1152, 2304, 4608, 4608, 4608, 4608, 8192, 6656, 3584, 8192, 6656, 3584, 1600 };

  // quantize + pack weights into DMA-friendly [kb][co][32] layout
  _Float16* wq[15];
  for (int i = 0; i < 15; ++i) {
    int Kpad = (wKt[i] + 31) & ~31;
    int n = wCo[i] * Kpad;
    wq[i] = allocH((size_t)n + 1024);  // +2KB slack for tail-tile overreads
    quant_w_pack_kernel<<<(n + 255) / 256, 256, 0, stream>>>(w[i], wq[i], wCo[i], wKt[i], Kpad);
  }

  const int B = 8;
  _Float16* c1   = allocH((size_t)B *  64 * 128 * 128);
  _Float16* c2   = allocH((size_t)B * 128 *  64 *  64);
  _Float16* c3   = allocH((size_t)B * 256 *  32 *  32);
  _Float16* c4   = allocH((size_t)B * 512 *  16 *  16);
  _Float16* r11  = allocH((size_t)B * 512 *  16 *  16);
  _Float16* r12  = allocH((size_t)B * 512 *  16 *  16);
  _Float16* r21  = allocH((size_t)B * 512 *  16 *  16);
  _Float16* r22  = allocH((size_t)B * 512 *  16 *  16);
  _Float16* f4   = allocH((size_t)B *  32 *  32 *  32);
  _Float16* d3   = allocH((size_t)B * 128 *  32 *  32);
  _Float16* cat3 = allocH((size_t)B * 416 *  32 *  32);
  _Float16* f3   = allocH((size_t)B *  32 *  64 *  64);
  _Float16* d2   = allocH((size_t)B *  64 *  64 *  64);
  _Float16* cat2 = allocH((size_t)B * 224 *  64 *  64);
  _Float16* f2   = allocH((size_t)B *  32 * 128 * 128);
  _Float16* d1   = allocH((size_t)B *   4 * 128 * 128);
  _Float16* cat1 = allocH((size_t)B * 100 * 128 * 128);
  _Float16* u10  = allocH((size_t)B *  32 * 257 * 257);
  float*    feat = allocF((size_t)B * 32);

  auto conv = [&](const void* xin, int xf32, const _Float16* wgt, const _Float16* res,
                  _Float16* out, int Ci, int Hi, int Wi, int Co, int Kh, int Kw,
                  int s, int pd, int Ho, int Wo, int ep) {
    dim3 g((Co + 31) / 32, (Ho * Wo + 255) / 256, B);
    qconv_wmma_kernel<<<g, 256, 0, stream>>>(xin, xf32, wgt, res, out,
                                             Ci, Hi, Wi, Co, Kh, Kw, s, pd, Ho, Wo, ep);
  };

  // encoder
  conv(x,   1, wq[0],  nullptr, c1,   4, 256, 256,  64, 3, 3, 2,  1, 128, 128, 0);
  conv(c1,  0, wq[1],  nullptr, c2,  64, 128, 128, 128, 3, 3, 2,  1,  64,  64, 0);
  conv(c2,  0, wq[2],  nullptr, c3, 128,  64,  64, 256, 3, 3, 2,  1,  32,  32, 0);
  conv(c3,  0, wq[3],  nullptr, c4, 256,  32,  32, 512, 3, 3, 2,  1,  16,  16, 0);
  // residual gates: r = (1 - q(conv)) * skip   (fused epilogue)
  conv(c4,  0, wq[4],  nullptr, r11, 512, 16, 16, 512, 3, 3, 1,  1, 16, 16, 0);
  conv(r11, 0, wq[5],  c4,      r12, 512, 16, 16, 512, 3, 3, 1,  1, 16, 16, 1);
  conv(r12, 0, wq[6],  nullptr, r21, 512, 16, 16, 512, 3, 3, 1,  1, 16, 16, 0);
  conv(r21, 0, wq[7],  r12,     r22, 512, 16, 16, 512, 3, 3, 1,  1, 16, 16, 1);
  // decoder stage 3: pad_hw(.,8)+pad2 -> eff pad 10; crop 33->32
  conv(r22, 0, wq[11], nullptr, f4,  512, 16, 16,  32, 4, 4, 1, 10, 32, 32, 0);
  conv(r22, 0, wq[8],  nullptr, d3,  512, 16, 16, 128, 4, 4, 1, 10, 32, 32, 0);
  { long t = (long)B * 416 * 32 * 32;
    concat3_kernel<<<(t + 255) / 256, 256, 0, stream>>>(c3, 256, d3, 128, f4, 32, cat3, B, 32 * 32); }
  // stage 2: eff pad 18; crop 65->64
  conv(cat3, 0, wq[12], nullptr, f3, 416, 32, 32,  32, 4, 4, 1, 18, 64, 64, 0);
  conv(cat3, 0, wq[9],  nullptr, d2, 416, 32, 32,  64, 4, 4, 1, 18, 64, 64, 0);
  { long t = (long)B * 224 * 64 * 64;
    concat3_kernel<<<(t + 255) / 256, 256, 0, stream>>>(c2, 128, d2, 64, f3, 32, cat2, B, 64 * 64); }
  // stage 1: eff pad 34; crop 129->128
  conv(cat2, 0, wq[13], nullptr, f2, 224, 64, 64,  32, 4, 4, 1, 34, 128, 128, 0);
  conv(cat2, 0, wq[10], nullptr, d1, 224, 64, 64,   4, 4, 4, 1, 34, 128, 128, 0);
  { long t = (long)B * 100 * 128 * 128;
    concat3_kernel<<<(t + 255) / 256, 256, 0, stream>>>(c1, 64, d1, 4, f2, 32, cat1, B, 128 * 128); }
  // final: eff pad 66 -> 257x257, no crop
  conv(cat1, 0, wq[14], nullptr, u10, 100, 128, 128, 32, 4, 4, 1, 66, 257, 257, 0);

  maxpool_kernel<<<B * 32, 256, 0, stream>>>(u10, feat, 257 * 257);
  fc_kernel<<<1, 32, 0, stream>>>(feat, (const float*)d_in[16], (float*)d_out);
}
